// MnistNet_59493886984571
// MI455X (gfx1250) — matrix operand order repaired and checked by reading, not compiled, not run on hardware
//
#include <hip/hip_runtime.h>
#include <math.h>

// ---------------------------------------------------------------------------
// CDNA5 (gfx1250) ternary-MLP forward.
// GEMMs: V_WMMA_F32_16X16X32_F16 (wave32), f16 operands / f32 accumulate,
// fed by GLOBAL_LOAD_ASYNC_TO_LDS_B128 into a double-buffered LDS pipeline
// (ASYNCcnt-tracked, s_wait_asynccnt). BatchNorm (training mode) forces
// per-layer materialization: ternarize(W)->f16, GEMM, column stats, BN+clip.
// ---------------------------------------------------------------------------

typedef _Float16 h16;
typedef __attribute__((ext_vector_type(16))) _Float16 v16h;
typedef __attribute__((ext_vector_type(8)))  float    v8f;

#define BM 128
#define BN 128
#define BK 32
#define LDSS 40                    // LDS row stride in halves (80B)
#define TILE_BYTES (BM * LDSS * 2) // 10240 B per A or B tile
#define STAGE_BYTES (2 * TILE_BYTES)

// Async direct memory->LDS copy of 16 bytes (CDNA5 VGLOBAL async op).
__device__ __forceinline__ void async_cp16(unsigned lds_off, const void* gptr)
{
  asm volatile("global_load_async_to_lds_b128 %0, %1, off"
               :: "v"(lds_off), "v"(gptr) : "memory");
}

__device__ __forceinline__ void wait_async0()
{
  asm volatile("s_wait_asynccnt 0x0" ::: "memory");
}

// ---------------------------------------------------------------------------
// Main WMMA GEMM:  Y[M,N] = X[M,K] @ W[N,K]^T + bias   (f16 in, f32 out)
// Block = 128x128, 256 threads (8 waves); wave grid 4(M) x 2(N); each wave
// computes 2x4 tiles of 16x16 (v8f accumulators). Double-buffered LDS.
// ---------------------------------------------------------------------------
__global__ void __launch_bounds__(256) gemm_f16_wmma(
    const h16* __restrict__ X, const h16* __restrict__ W,
    const float* __restrict__ bias, float* __restrict__ Y,
    int M, int N, int K)
{
  __shared__ h16 smem[2 * 2 * BM * LDSS];   // [stage][A|B][128][LDSS]
  const unsigned lds0 = (unsigned)(uintptr_t)&smem[0]; // LDS byte offset

  const int tid  = threadIdx.x;
  const int lane = tid & 31;
  const int wid  = tid >> 5;     // 0..7
  const int wm   = wid >> 1;     // 0..3 -> 32-row strip
  const int wn   = wid & 1;      // 0..1 -> 64-col strip

  const int m0 = blockIdx.y * BM;
  const int n0 = blockIdx.x * BN;

  v8f acc[2][4] = {};

  const int r  = lane & 15;      // row (A) / column (B) within tile
  const int hk = lane >> 4;      // lane-half selects K sub-chunk

  // Issue one K-tile (A: 128x32, B: 128x32) as 4 async b128 copies/thread.
  auto issue = [&](int kk, int st) {
    unsigned base = lds0 + (unsigned)(st * STAGE_BYTES);
    #pragma unroll
    for (int c = 0; c < 2; ++c) {
      int chunk = tid + (c << 8);        // 0..511
      int row   = chunk >> 2;
      int colh  = (chunk & 3) << 3;      // halves
      unsigned loff = (unsigned)(row * LDSS + colh) * 2u;
      async_cp16(base + loff,
                 &X[(size_t)(m0 + row) * K + kk + colh]);
      async_cp16(base + TILE_BYTES + loff,
                 &W[(size_t)(n0 + row) * K + kk + colh]);
    }
  };

  issue(0, 0);                   // prologue: fill stage 0

  int s = 0;
  for (int k0 = 0; k0 < K; k0 += BK, s ^= 1) {
    wait_async0();               // my async writes into stage s have landed
    __syncthreads();             // all waves: writes landed, prior reads done

    if (k0 + BK < K) {
      issue(k0 + BK, s ^ 1);     // async engine fills next stage
      if (k0 + 2 * BK < K) {     // warm L2 two tiles ahead (global_prefetch_b8)
        __builtin_prefetch(&X[(size_t)(m0 + (tid >> 1)) * K + k0 + 2 * BK], 0, 3);
        __builtin_prefetch(&W[(size_t)(n0 + (tid >> 1)) * K + k0 + 2 * BK], 0, 3);
      }
    }

    const h16* sA = smem + s * (2 * BM * LDSS);
    const h16* sB = sA + BM * LDSS;

    // A fragments (16x32 f16, ISA 7.12.2): lane<16 holds K 0..7 & 16..23,
    // lane>=16 holds K 8..15 & 24..31, for row r.
    v16h a[2];
    #pragma unroll
    for (int mi = 0; mi < 2; ++mi) {
      const h16* p = &sA[(wm * 32 + mi * 16 + r) * LDSS + hk * 8];
      uint4* d = reinterpret_cast<uint4*>(&a[mi]);
      d[0] = *(const uint4*)(p);
      d[1] = *(const uint4*)(p + 16);
    }
    // B fragments (32x16 f16): lane<16 holds K 0..7 & 16..23, lane>=16 holds K 8..15 & 24..31 of col r.
    v16h b[4];
    #pragma unroll
    for (int ni = 0; ni < 4; ++ni) {
      const h16* p = &sB[(wn * 64 + ni * 16 + r) * LDSS + hk * 8];
      uint4* d = reinterpret_cast<uint4*>(&b[ni]);
      d[0] = *(const uint4*)(p);
      d[1] = *(const uint4*)(p + 16);
    }

    #pragma unroll
    for (int mi = 0; mi < 2; ++mi)
      #pragma unroll
      for (int ni = 0; ni < 4; ++ni)
        acc[mi][ni] = __builtin_amdgcn_wmma_f32_16x16x32_f16(
            false, a[mi], false, b[ni], (short)0, acc[mi][ni], false, false);
  }

  // Epilogue: C layout (VGPR j, lane l) -> M = j + (l>>4)*8, N = l&15.
  #pragma unroll
  for (int mi = 0; mi < 2; ++mi) {
    #pragma unroll
    for (int ni = 0; ni < 4; ++ni) {
      int col  = n0 + wn * 64 + ni * 16 + (lane & 15);
      float bv = bias[col];
      #pragma unroll
      for (int j = 0; j < 8; ++j) {
        int row = m0 + wm * 32 + mi * 16 + j + (lane >> 4) * 8;
        Y[(size_t)row * N + col] = acc[mi][ni][j] + bv;
      }
    }
  }
}

// ---------------------------------------------------------------------------
// Helpers: f32->f16 pad, deterministic |W| reduction, ternarize, BN, etc.
// ---------------------------------------------------------------------------
__global__ void f32_to_f16_pad(const float* __restrict__ src, h16* __restrict__ dst,
                               int M, int K, int Kp)
{
  size_t total = (size_t)M * Kp;
  size_t step  = (size_t)gridDim.x * blockDim.x;
  for (size_t i = (size_t)blockIdx.x * blockDim.x + threadIdx.x; i < total; i += step) {
    int k = (int)(i % Kp);
    size_t row = i / Kp;
    float v = (k < K) ? src[row * K + k] : 0.f;
    dst[i] = (h16)v;
  }
}

__global__ void abs_sum_partial(const float* __restrict__ W, size_t n,
                                float* __restrict__ partials)
{
  __shared__ float s[256];
  float a = 0.f;
  size_t step = (size_t)gridDim.x * 256;
  for (size_t i = (size_t)blockIdx.x * 256 + threadIdx.x; i < n; i += step)
    a += fabsf(W[i]);
  s[threadIdx.x] = a;
  __syncthreads();
  for (int o = 128; o > 0; o >>= 1) {
    if ((int)threadIdx.x < o) s[threadIdx.x] += s[threadIdx.x + o];
    __syncthreads();
  }
  if (threadIdx.x == 0) partials[blockIdx.x] = s[0];
}

__global__ void reduce_1024(const float* __restrict__ partials, float* __restrict__ out)
{
  __shared__ float s[256];
  float a = partials[threadIdx.x] + partials[threadIdx.x + 256]
          + partials[threadIdx.x + 512] + partials[threadIdx.x + 768];
  s[threadIdx.x] = a;
  __syncthreads();
  for (int o = 128; o > 0; o >>= 1) {
    if ((int)threadIdx.x < o) s[threadIdx.x] += s[threadIdx.x + o];
    __syncthreads();
  }
  if (threadIdx.x == 0) *out = s[0];
}

// W (N,K) f32 -> dst (N,Kp) f16 ternary {-1,0,+1}, threshold 0.7*mean|W|.
__global__ void ternarize_f16(const float* __restrict__ W, h16* __restrict__ dst,
                              int N, int K, int Kp,
                              const float* __restrict__ sumabs, float inv_count)
{
  float thr = 0.7f * (*sumabs) * inv_count;
  size_t total = (size_t)N * Kp;
  size_t step  = (size_t)gridDim.x * blockDim.x;
  for (size_t i = (size_t)blockIdx.x * blockDim.x + threadIdx.x; i < total; i += step) {
    int k = (int)(i % Kp);
    size_t n = i / Kp;
    float w = (k < K) ? W[n * K + k] : 0.f;
    float t = (fabsf(w) > thr) ? ((w > 0.f) ? 1.f : -1.f) : 0.f;
    dst[i] = (h16)t;
  }
}

// Column mean / rstd over the batch dim (training-mode BN statistics).
__global__ void colstats(const float* __restrict__ Y, int M, int N,
                         float* __restrict__ mean, float* __restrict__ rstd)
{
  __shared__ float ss[4][64];
  __shared__ float sq[4][64];
  int tx = threadIdx.x, ty = threadIdx.y;
  int n = blockIdx.x * 64 + tx;
  float a = 0.f, b = 0.f;
  if (n < N) {
    for (int m = ty; m < M; m += 4) {
      float v = Y[(size_t)m * N + n];
      a += v; b += v * v;
    }
  }
  ss[ty][tx] = a; sq[ty][tx] = b;
  __syncthreads();
  if (ty == 0 && n < N) {
    float A = ss[0][tx] + ss[1][tx] + ss[2][tx] + ss[3][tx];
    float B = sq[0][tx] + sq[1][tx] + sq[2][tx] + sq[3][tx];
    float mu = A / (float)M;
    float var = B / (float)M - mu * mu;
    mean[n] = mu;
    rstd[n] = rsqrtf(var + 1e-5f);
  }
}

// BN + hardtanh + convert to f16 (next layer input).
__global__ void bn_apply_f16(const float* __restrict__ Y,
                             const float* __restrict__ mean, const float* __restrict__ rstd,
                             const float* __restrict__ gamma, const float* __restrict__ beta,
                             h16* __restrict__ Xout, size_t total, int N)
{
  size_t step = (size_t)gridDim.x * blockDim.x;
  for (size_t i = (size_t)blockIdx.x * blockDim.x + threadIdx.x; i < total; i += step) {
    int n = (int)(i % N);
    float v = gamma[n] * (Y[i] - mean[n]) * rstd[n] + beta[n];
    v = fminf(1.f, fmaxf(-1.f, v));
    Xout[i] = (h16)v;
  }
}

// Tiny output layer (N=10): one thread per (row,col) dot product.
__global__ void out_gemm(const h16* __restrict__ X, const h16* __restrict__ W,
                         const float* __restrict__ bias, float* __restrict__ Yout,
                         int M, int K)
{
  int idx = blockIdx.x * blockDim.x + threadIdx.x;
  if (idx >= M * 10) return;
  int row = idx / 10, col = idx % 10;
  const h16* xp = X + (size_t)row * K;
  const h16* wp = W + (size_t)col * K;
  float acc = 0.f;
  #pragma unroll 4
  for (int k = 0; k < K; k += 2)
    acc += (float)xp[k] * (float)wp[k] + (float)xp[k + 1] * (float)wp[k + 1];
  Yout[idx] = acc + bias[col];
}

// Final BN + log_softmax over 10 classes; one thread per row.
__global__ void bn_logsoftmax(const float* __restrict__ Yout,
                              const float* __restrict__ mean, const float* __restrict__ rstd,
                              const float* __restrict__ g, const float* __restrict__ b,
                              float* __restrict__ out, int M)
{
  int row = blockIdx.x * blockDim.x + threadIdx.x;
  if (row >= M) return;
  float z[10];
  float mx = -1e30f;
  #pragma unroll
  for (int c = 0; c < 10; ++c) {
    float v = g[c] * (Yout[row * 10 + c] - mean[c]) * rstd[c] + b[c];
    z[c] = v;
    mx = fmaxf(mx, v);
  }
  float s = 0.f;
  #pragma unroll
  for (int c = 0; c < 10; ++c) s += expf(z[c] - mx);
  float lse = mx + logf(s);
  #pragma unroll
  for (int c = 0; c < 10; ++c) out[row * 10 + c] = z[c] - lse;
}

// ---------------------------------------------------------------------------
extern "C" void kernel_launch(void* const* d_in, const int* in_sizes, int n_in,
                              void* d_out, int out_size, void* d_ws, size_t ws_size,
                              hipStream_t stream)
{
  const float* x     = (const float*)d_in[0];   // 8192 x 784
  const float* W_in  = (const float*)d_in[1];   // 4096 x 784
  const float* b_in  = (const float*)d_in[2];   // 4096
  const float* W_hid = (const float*)d_in[3];   // 4 x 4096 x 4096
  const float* b_hid = (const float*)d_in[4];   // 4 x 4096
  const float* W_out = (const float*)d_in[5];   // 10 x 4096
  const float* b_out = (const float*)d_in[6];   // 10
  const float* gamma = (const float*)d_in[7];   // 5 x 4096
  const float* beta  = (const float*)d_in[8];   // 5 x 4096
  const float* g_out = (const float*)d_in[9];   // 10
  const float* be_out= (const float*)d_in[10];  // 10
  float* out = (float*)d_out;                   // 8192 x 10

  const int M = 8192, H = 4096;

  // Workspace carve-up (all chunks 256B aligned by construction).
  char* w = (char*)d_ws;
  h16*  Xa = (h16*)w;            w += (size_t)M * H * 2;     // 64 MiB
  h16*  Xb = (h16*)w;            w += (size_t)M * H * 2;     // 64 MiB
  float* Y = (float*)w;          w += (size_t)M * H * 4;     // 128 MiB
  h16*  Wf = (h16*)w;            w += (size_t)H * H * 2;     // 32 MiB
  float* mean     = (float*)w;   w += H * 4;
  float* rstd     = (float*)w;   w += H * 4;
  float* partials = (float*)w;   w += 1024 * 4;
  float* sumabs   = (float*)w;   w += 256;

  auto gemm = [&](const h16* Xp, const h16* Wp, const float* bp, float* Yp,
                  int m, int n, int k) {
    dim3 grid(n / BN, m / BM);
    gemm_f16_wmma<<<grid, 256, 0, stream>>>(Xp, Wp, bp, Yp, m, n, k);
  };

  // ---- Layer 0: 784 padded to 800 (25 K-steps of 32) ----------------------
  f32_to_f16_pad<<<2048, 256, 0, stream>>>(x, Xa, M, 784, 800);

  abs_sum_partial<<<1024, 256, 0, stream>>>(W_in, (size_t)H * 784, partials);
  reduce_1024<<<1, 256, 0, stream>>>(partials, sumabs);
  ternarize_f16<<<2048, 256, 0, stream>>>(W_in, Wf, H, 784, 800, sumabs,
                                          1.f / ((float)H * 784.f));
  gemm(Xa, Wf, b_in, Y, M, H, 800);
  colstats<<<H / 64, dim3(64, 4), 0, stream>>>(Y, M, H, mean, rstd);
  bn_apply_f16<<<4096, 256, 0, stream>>>(Y, mean, rstd, gamma, beta, Xb,
                                         (size_t)M * H, H);

  // ---- 4 hidden layers ----------------------------------------------------
  h16* cur = Xb;
  h16* nxt = Xa;
  for (int i = 0; i < 4; ++i) {
    const float* Wi = W_hid + (size_t)i * H * H;
    abs_sum_partial<<<1024, 256, 0, stream>>>(Wi, (size_t)H * H, partials);
    reduce_1024<<<1, 256, 0, stream>>>(partials, sumabs);
    ternarize_f16<<<4096, 256, 0, stream>>>(Wi, Wf, H, H, H, sumabs,
                                            1.f / ((float)H * (float)H));
    gemm(cur, Wf, b_hid + (size_t)i * H, Y, M, H, H);
    colstats<<<H / 64, dim3(64, 4), 0, stream>>>(Y, M, H, mean, rstd);
    bn_apply_f16<<<4096, 256, 0, stream>>>(Y, mean, rstd,
                                           gamma + (size_t)(i + 1) * H,
                                           beta + (size_t)(i + 1) * H,
                                           nxt, (size_t)M * H, H);
    h16* t = cur; cur = nxt; nxt = t;
  }

  // ---- Output layer (N=10) + BN + log_softmax -----------------------------
  abs_sum_partial<<<1024, 256, 0, stream>>>(W_out, (size_t)10 * H, partials);
  reduce_1024<<<1, 256, 0, stream>>>(partials, sumabs);
  ternarize_f16<<<160, 256, 0, stream>>>(W_out, Wf, 10, H, H, sumabs,
                                         1.f / (10.f * (float)H));
  out_gemm<<<(M * 10 + 255) / 256, 256, 0, stream>>>(cur, Wf, b_out, Y, M, H);
  colstats<<<1, dim3(64, 4), 0, stream>>>(Y, M, 10, mean, rstd);
  bn_logsoftmax<<<(M + 255) / 256, 256, 0, stream>>>(Y, mean, rstd, g_out,
                                                     be_out, out, M);
}
